// DendriticANN_21646635172242
// MI455X (gfx1250) — matrix operand order repaired and checked by reading, not compiled, not run on hardware
//
#include <hip/hip_runtime.h>
#include <stdint.h>

// ---------------------------------------------------------------------------
// DendriticANN forward, MI455X (gfx1250, wave32, WMMA bf16 -> f32 accum)
// All f32->bf16 conversion hoisted into one memory-bound pass so the WMMA
// loops contain only 16B loads + v_wmma (no per-element VALU conversion).
// ---------------------------------------------------------------------------

typedef __attribute__((ext_vector_type(16))) __bf16   v16bf;
typedef __attribute__((ext_vector_type(8)))  __bf16   v8bf;
typedef __attribute__((ext_vector_type(8)))  float    v8f;
typedef __attribute__((ext_vector_type(8)))  uint16_t v8u16;

#define B_N   2048
#define IN_N  1024
#define H_N   512
#define D_N   32
#define OUT_N 10
#define L_N   2

__device__ __forceinline__ float leaky(float v) {
    return v >= 0.f ? v : 0.01f * v;
}

// f32 -> bf16 round-to-nearest-even
__device__ __forceinline__ uint16_t f2bf_bits(float f) {
    uint32_t u = __builtin_bit_cast(uint32_t, f);
    u += 0x7FFFu + ((u >> 16) & 1u);
    return (uint16_t)(u >> 16);
}

__device__ __forceinline__ v16bf concat8(v8bf lo, v8bf hi) {
    return __builtin_shufflevector(lo, hi, 0, 1, 2, 3, 4, 5, 6, 7,
                                           8, 9, 10, 11, 12, 13, 14, 15);
}

// ---------------------------------------------------------------------------
// Kernel 0: bulk f32 -> bf16 (8 elements / thread, 16B stores)
// ---------------------------------------------------------------------------
__global__ void __launch_bounds__(256)
cvt_bf16_kernel(const float* __restrict__ src, uint16_t* __restrict__ dst, int n8) {
    const int i = blockIdx.x * 256 + threadIdx.x;
    if (i >= n8) return;
    const float4* p = (const float4*)src + 2 * (size_t)i;
    const float4 a = p[0], b = p[1];
    v8u16 o;
    o[0] = f2bf_bits(a.x); o[1] = f2bf_bits(a.y);
    o[2] = f2bf_bits(a.z); o[3] = f2bf_bits(a.w);
    o[4] = f2bf_bits(b.x); o[5] = f2bf_bits(b.y);
    o[6] = f2bf_bits(b.z); o[7] = f2bf_bits(b.w);
    *(v8u16*)(dst + 8 * (size_t)i) = o;
}

// ---------------------------------------------------------------------------
// Kernel 1: hA = leaky(x @ W_in^T + b_in)        [B,IN]x[IN,H] -> [B,H]
// bf16 operands from workspace. 1 wave = 16x16 tile; block = 8 waves.
// grid = (B/16, H/128)
// ---------------------------------------------------------------------------
__global__ void __launch_bounds__(256)
gemm_in_wmma_kernel(const uint16_t* __restrict__ x_bf,
                    const uint16_t* __restrict__ Win_bf,
                    const float* __restrict__ b_in, float* __restrict__ hout) {
    const int lane = threadIdx.x & 31;
    const int wave = threadIdx.x >> 5;
    const int half = lane >> 4;
    const int l15  = lane & 15;
    const int m0 = blockIdx.x * 16;
    const int n0 = (blockIdx.y * 8 + wave) * 16;

    const __bf16* arow = (const __bf16*)x_bf   + (size_t)(m0 + l15) * IN_N;
    const __bf16* brow = (const __bf16*)Win_bf + (size_t)(n0 + l15) * IN_N;

    v8f acc = {};
    for (int k0 = 0; k0 < IN_N; k0 += 32) {
        // A: K = k0 + 16h + {0..15}
        const v8bf a0 = *(const v8bf*)(arow + k0 + 16 * half);
        const v8bf a1 = *(const v8bf*)(arow + k0 + 16 * half + 8);
        // B: K = k0 + 16h + {0..15}
        const v8bf b0 = *(const v8bf*)(brow + k0 + 16 * half);
        const v8bf b1 = *(const v8bf*)(brow + k0 + 16 * half + 8);
        acc = __builtin_amdgcn_wmma_f32_16x16x32_bf16(false, concat8(a0, a1),
                                                      false, concat8(b0, b1),
                                                      (short)0, acc, false, false);
    }

    const int   col  = n0 + l15;
    const float bias = b_in[col];
    #pragma unroll
    for (int v = 0; v < 8; ++v) {
        const int row = m0 + v + 8 * half;
        hout[(size_t)row * H_N + col] = leaky(acc[v] + bias);
    }
}

// ---------------------------------------------------------------------------
// Kernel 2: per-column batch stats (mean, rstd).  grid = H, block = 256
// ---------------------------------------------------------------------------
__global__ void __launch_bounds__(256)
bn_stats_kernel(const float* __restrict__ h, float* __restrict__ stats) {
    __shared__ float ssum[256];
    __shared__ float ssq[256];
    const int c = blockIdx.x, tid = threadIdx.x;
    float s = 0.f, q = 0.f;
    for (int r = tid; r < B_N; r += 256) {
        float v = h[(size_t)r * H_N + c];
        s += v; q += v * v;
    }
    ssum[tid] = s; ssq[tid] = q;
    __syncthreads();
    for (int st = 128; st > 0; st >>= 1) {
        if (tid < st) { ssum[tid] += ssum[tid + st]; ssq[tid] += ssq[tid + st]; }
        __syncthreads();
    }
    if (tid == 0) {
        float mean = ssum[0] * (1.f / B_N);
        float var  = ssq[0] * (1.f / B_N) - mean * mean;
        stats[c]       = mean;
        stats[H_N + c] = rsqrtf(var + 1e-5f);
    }
}

// ---------------------------------------------------------------------------
// Kernel 3: hout = (hin - mean) * rstd            grid = B*H/256
// ---------------------------------------------------------------------------
__global__ void __launch_bounds__(256)
bn_apply_kernel(const float* __restrict__ hin, const float* __restrict__ stats,
                float* __restrict__ hout) {
    const int idx = blockIdx.x * 256 + threadIdx.x;
    const int c = idx & (H_N - 1);
    hout[idx] = (hin[idx] - stats[c]) * stats[H_N + c];
}

// ---------------------------------------------------------------------------
// Kernel 4: row L2-normalize + cast to bf16.      grid = B, block = 256
// ---------------------------------------------------------------------------
__global__ void __launch_bounds__(256)
row_norm_bf16_kernel(const float* __restrict__ h, uint16_t* __restrict__ xn) {
    __shared__ float red[256];
    const int b = blockIdx.x, tid = threadIdx.x;
    const float* row = h + (size_t)b * H_N;
    float s = 0.f;
    for (int i = tid; i < H_N; i += 256) { float v = row[i]; s += v * v; }
    red[tid] = s;
    __syncthreads();
    for (int st = 128; st > 0; st >>= 1) {
        if (tid < st) red[tid] += red[tid + st];
        __syncthreads();
    }
    const float scale = 1.f / fmaxf(sqrtf(red[0]), 1e-12f);
    for (int i = tid; i < H_N; i += 256)
        xn[(size_t)b * H_N + i] = f2bf_bits(row[i] * scale);
}

// ---------------------------------------------------------------------------
// Kernel 5: dendritic layer, fully fused (bf16 WMMA, f32 acc):
//   dend[b,n,d] = leaky( sum_i xn[b,i] * Wd[n,d,i] )
//   t[b,n]      = sum_d dend[b,n,d] * soma[n,d]          (shfl_xor reduce)
//   hout[b,n]   = leaky(leaky(t))
// 1 wave = 16 rows x 32 dendrites (2 accs). block = 8 waves = 128 rows.
// grid = (B/128, H)
// ---------------------------------------------------------------------------
__global__ void __launch_bounds__(256)
dendritic_wmma_kernel(const uint16_t* __restrict__ xn_u16,
                      const uint16_t* __restrict__ Wd_bf_l,
                      const float* __restrict__ soma_l,
                      float* __restrict__ hout) {
    const int lane = threadIdx.x & 31;
    const int wave = threadIdx.x >> 5;
    const int half = lane >> 4;
    const int l15  = lane & 15;
    const int m0     = blockIdx.x * 128 + wave * 16;
    const int neuron = blockIdx.y;

    const __bf16* arow = (const __bf16*)xn_u16 + (size_t)(m0 + l15) * H_N;
    const __bf16* w0   = (const __bf16*)Wd_bf_l + ((size_t)neuron * D_N + l15) * H_N;
    const __bf16* w1   = w0 + (size_t)16 * H_N;

    v8f acc0 = {};
    v8f acc1 = {};
    for (int k0 = 0; k0 < H_N; k0 += 32) {
        if (k0 + 32 < H_N) {
            __builtin_prefetch(w0 + k0 + 32 + 16 * half, 0, 0);
            __builtin_prefetch(w1 + k0 + 32 + 16 * half, 0, 0);
        }
        const v8bf a0 = *(const v8bf*)(arow + k0 + 16 * half);
        const v8bf a1 = *(const v8bf*)(arow + k0 + 16 * half + 8);
        const v16bf a = concat8(a0, a1);

        const v8bf p0 = *(const v8bf*)(w0 + k0 + 16 * half);
        const v8bf p1 = *(const v8bf*)(w0 + k0 + 16 * half + 8);
        const v8bf q0 = *(const v8bf*)(w1 + k0 + 16 * half);
        const v8bf q1 = *(const v8bf*)(w1 + k0 + 16 * half + 8);

        acc0 = __builtin_amdgcn_wmma_f32_16x16x32_bf16(false, a, false, concat8(p0, p1),
                                                       (short)0, acc0, false, false);
        acc1 = __builtin_amdgcn_wmma_f32_16x16x32_bf16(false, a, false, concat8(q0, q1),
                                                       (short)0, acc1, false, false);
    }

    // fused soma reduction + activations
    const float s_lo = soma_l[neuron * D_N + l15];
    const float s_hi = soma_l[neuron * D_N + 16 + l15];
    #pragma unroll
    for (int v = 0; v < 8; ++v) {
        float t = leaky(acc0[v]) * s_lo + leaky(acc1[v]) * s_hi;
        t += __shfl_xor(t, 8, 32);
        t += __shfl_xor(t, 4, 32);
        t += __shfl_xor(t, 2, 32);
        t += __shfl_xor(t, 1, 32);
        if (l15 == v) {
            const int row = m0 + v + 8 * half;
            hout[(size_t)row * H_N + neuron] = leaky(leaky(t));
        }
    }
}

// ---------------------------------------------------------------------------
// Kernel 6: out = h @ W_out^T + b_out             [B,H]x[H,OUT] -> [B,OUT]
// ---------------------------------------------------------------------------
__global__ void __launch_bounds__(64)
out_gemm_kernel(const float* __restrict__ h, const float* __restrict__ W_out,
                const float* __restrict__ b_out, float* __restrict__ out) {
    __shared__ float row[H_N];
    const int b = blockIdx.x, tid = threadIdx.x;
    for (int i = tid; i < H_N; i += 64) row[i] = h[(size_t)b * H_N + i];
    __syncthreads();
    if (tid < OUT_N) {
        float acc = b_out[tid];
        const float* w = W_out + (size_t)tid * H_N;
        for (int k = 0; k < H_N; ++k) acc += row[k] * w[k];
        out[b * OUT_N + tid] = acc;
    }
}

// ---------------------------------------------------------------------------
// Launch
// ---------------------------------------------------------------------------
extern "C" void kernel_launch(void* const* d_in, const int* in_sizes, int n_in,
                              void* d_out, int out_size, void* d_ws, size_t ws_size,
                              hipStream_t stream) {
    const float* x     = (const float*)d_in[0];   // [B, IN]
    const float* W_in  = (const float*)d_in[1];   // [H, IN]
    const float* b_in  = (const float*)d_in[2];   // [H]
    const float* Wd    = (const float*)d_in[3];   // [L, H, D, H]
    const float* soma  = (const float*)d_in[4];   // [L, H, D]
    const float* W_out = (const float*)d_in[5];   // [OUT, H]
    const float* b_out = (const float*)d_in[6];   // [OUT]
    float*       out   = (float*)d_out;           // [B, OUT]

    char* ws = (char*)d_ws;
    float*    hA     = (float*)ws;                                // 4 MB
    float*    hB     = (float*)(ws + (size_t)4  * 1024 * 1024);   // 4 MB
    uint16_t* xn     = (uint16_t*)(ws + (size_t)8 * 1024 * 1024); // 2 MB
    float*    stats  = (float*)(ws + (size_t)10 * 1024 * 1024);   // 4 KB
    uint16_t* x_bf   = (uint16_t*)(ws + (size_t)11 * 1024 * 1024);// 4 MB
    uint16_t* Win_bf = (uint16_t*)(ws + (size_t)15 * 1024 * 1024);// 1 MB
    uint16_t* Wd_bf  = (uint16_t*)(ws + (size_t)16 * 1024 * 1024);// 32 MB

    // ---- one-shot bf16 conversion passes (memory bound, ~4 us total) ----
    {
        const int n8_x  = (B_N * IN_N) / 8;                     // 262144
        const int n8_wi = (H_N * IN_N) / 8;                     // 65536
        const int n8_wd = (L_N * H_N * D_N * H_N) / 8;          // 2097152
        cvt_bf16_kernel<<<(n8_x  + 255) / 256, 256, 0, stream>>>(x,    x_bf,   n8_x);
        cvt_bf16_kernel<<<(n8_wi + 255) / 256, 256, 0, stream>>>(W_in, Win_bf, n8_wi);
        cvt_bf16_kernel<<<(n8_wd + 255) / 256, 256, 0, stream>>>(Wd,   Wd_bf,  n8_wd);
    }

    // input layer: hA = leaky(x @ W_in^T + b_in)
    gemm_in_wmma_kernel<<<dim3(B_N / 16, H_N / 128), 256, 0, stream>>>(x_bf, Win_bf, b_in, hA);
    // hB = batchnorm(hA)
    bn_stats_kernel<<<H_N, 256, 0, stream>>>(hA, stats);
    bn_apply_kernel<<<(B_N * H_N) / 256, 256, 0, stream>>>(hA, stats, hB);

    for (int l = 0; l < L_N; ++l) {
        row_norm_bf16_kernel<<<B_N, 256, 0, stream>>>(hB, xn);
        const uint16_t* Wd_bf_l = Wd_bf + (size_t)l * H_N * D_N * H_N;
        const float*    soma_l  = soma  + (size_t)l * H_N * D_N;
        dendritic_wmma_kernel<<<dim3(B_N / 128, H_N), 256, 0, stream>>>(xn, Wd_bf_l, soma_l, hA);
        bn_stats_kernel<<<H_N, 256, 0, stream>>>(hA, stats);
        bn_apply_kernel<<<(B_N * H_N) / 256, 256, 0, stream>>>(hA, stats, hB);
    }

    out_gemm_kernel<<<B_N, 64, 0, stream>>>(hB, W_out, b_out, out);
}
